// RAREJEPA_46703474376740
// MI455X (gfx1250) — compile-verified, matrix operand order checked by reading, not run on hardware
//
#include <hip/hip_runtime.h>
#include <hip/hip_bf16.h>
#include <math.h>
#include <stdint.h>

// ---------------- model dims ----------------
#define DB   64
#define DL   512
#define DD   256
#define DV   32000
#define DK   6
#define DH   4
#define DHD  64
#define DFF  512
#define DM   (DB * DL)            // 32768 rows in the token stream

// ---------------- WMMA fragment types ----------------
typedef __attribute__((ext_vector_type(16))) __bf16        bfrag_t;
typedef __attribute__((ext_vector_type(8)))  float         facc_t;
typedef __attribute__((ext_vector_type(4)))  unsigned int  u32x4;

union FragU { bfrag_t f; u32x4 u[2]; };

// ---------------- CDNA5 async global->LDS path (ASYNCcnt) ----------------
#if __has_builtin(__builtin_amdgcn_global_load_async_to_lds_b128)
#define HAS_ASYNC_LDS 1
#else
#define HAS_ASYNC_LDS 0
#endif

#if HAS_ASYNC_LDS
// builtin signature (from hipcc diagnostic): (v4i AS1*, v4i AS3*, imm int, imm int)
typedef int v4i __attribute__((vector_size(16)));
typedef __attribute__((address_space(1))) v4i gv4i;
typedef __attribute__((address_space(3))) v4i lv4i;
// per-lane 16B DMA: global -> LDS, tracked by ASYNCcnt
__device__ __forceinline__ void async_cp16(const void* g, void* l) {
  __builtin_amdgcn_global_load_async_to_lds_b128(
      (gv4i*)(uintptr_t)g,
      (lv4i*)(unsigned int)(uintptr_t)l,   // low 32 bits of generic = LDS offset
      0, 0);
}
__device__ __forceinline__ void wait_async0() {
#if __has_builtin(__builtin_amdgcn_s_wait_asynccnt)
  __builtin_amdgcn_s_wait_asynccnt(0);
#else
  asm volatile("s_wait_asynccnt 0x0" ::: "memory");
#endif
}
#endif

// ---------------- reduction helpers ----------------
__device__ __forceinline__ float warpReduceSum(float v) {
#pragma unroll
  for (int m = 16; m >= 1; m >>= 1) v += __shfl_xor(v, m, 32);
  return v;
}
__device__ __forceinline__ float warpReduceMax(float v) {
#pragma unroll
  for (int m = 16; m >= 1; m >>= 1) v = fmaxf(v, __shfl_xor(v, m, 32));
  return v;
}
__device__ __forceinline__ float blockReduceSum256(float v, float* red) {
  v = warpReduceSum(v);
  __syncthreads();
  if ((threadIdx.x & 31) == 0) red[threadIdx.x >> 5] = v;
  __syncthreads();
  float t = 0.f;
#pragma unroll
  for (int i = 0; i < 8; ++i) t += red[i];
  return t;
}
__device__ __forceinline__ float blockReduceMax256(float v, float* red) {
  v = warpReduceMax(v);
  __syncthreads();
  if ((threadIdx.x & 31) == 0) red[threadIdx.x >> 5] = v;
  __syncthreads();
  float t = -3.4e38f;
#pragma unroll
  for (int i = 0; i < 8; ++i) t = fmaxf(t, red[i]);
  return t;
}

// ---------------- f32 -> bf16 convert ----------------
__global__ __launch_bounds__(256) void cvt_bf16_kernel(const float* __restrict__ in,
                                                       __bf16* __restrict__ out,
                                                       long long n) {
  long long stride = (long long)gridDim.x * 256;
  for (long long i = blockIdx.x * 256LL + threadIdx.x; i < n; i += stride)
    out[i] = (__bf16)in[i];
}

// ---------------- embedding + LayerNorm (fp32 residual stream) ----------------
__global__ __launch_bounds__(256) void embed_ln_kernel(const int* __restrict__ tokens,
                                                       const float* __restrict__ te,
                                                       const float* __restrict__ pe,
                                                       const float* __restrict__ g,
                                                       const float* __restrict__ b,
                                                       float* __restrict__ x) {
  __shared__ float red[8];
  const int row = blockIdx.x, t = threadIdx.x;
  const int l = row & (DL - 1);
  const int tok = tokens[row];
  float v = te[(size_t)tok * DD + t] + pe[(size_t)l * DD + t];
  const float mu = blockReduceSum256(v, red) * (1.f / DD);
  const float d = v - mu;
  const float var = blockReduceSum256(d * d, red) * (1.f / DD);
  x[(size_t)row * DD + t] = d * rsqrtf(var + 1e-5f) * g[t] + b[t];
}

// ---------------- LayerNorm of residual stream -> bf16 ----------------
__global__ __launch_bounds__(256) void ln_bf16_kernel(const float* __restrict__ x,
                                                      const float* __restrict__ g,
                                                      const float* __restrict__ b,
                                                      __bf16* __restrict__ h) {
  __shared__ float red[8];
  const int row = blockIdx.x, t = threadIdx.x;
  float v = x[(size_t)row * DD + t];
  const float mu = blockReduceSum256(v, red) * (1.f / DD);
  const float d = v - mu;
  const float var = blockReduceSum256(d * d, red) * (1.f / DD);
  h[(size_t)row * DD + t] = (__bf16)(d * rsqrtf(var + 1e-5f) * g[t] + b[t]);
}

// ---------------- WMMA GEMM: C[M,N] = A[M,K] * W[N,K]^T + bias ----------------
// MODE 0: Cf = v ; MODE 1: Cf += v (residual) ; MODE 2: Cb = bf16(v) ;
// MODE 3: Cb = bf16(gelu(v))
#define BM 128
#define BN 64
#define BK 32
#define LSTR 40   // padded LDS row stride (bf16 elems): 80B, conflict-free

#if HAS_ASYNC_LDS
// issue the 3 per-thread 16B DMAs for one (A,W) K-tile
__device__ __forceinline__ void stage_tile_async(
    const __bf16* __restrict__ A, const __bf16* __restrict__ W,
    int M, int lda, int ldw, int bm, int bn, int k0, int tid,
    __bf16* a_dst, __bf16* b_dst) {
#pragma unroll
  for (int i = 0; i < 2; ++i) {
    int c = tid + i * 256;
    int row = c >> 2, cc = (c & 3) * 8;
    int grow = bm * BM + row;
    if (grow >= M) grow = M - 1;   // clamp: garbage rows are masked in epilogue
    async_cp16(A + (size_t)grow * lda + k0 + cc, a_dst + row * LSTR + cc);
  }
  {
    int row = tid >> 2, cc = (tid & 3) * 8;
    async_cp16(W + (size_t)(bn * BN + row) * ldw + k0 + cc,
               b_dst + row * LSTR + cc);
  }
}
#endif

template <int MODE>
__global__ __launch_bounds__(256) void gemm_wmma_kernel(
    const __bf16* __restrict__ A, const __bf16* __restrict__ W,
    const float* __restrict__ bias,
    float* __restrict__ Cf, __bf16* __restrict__ Cb,
    int M, int N, int Kd, int lda, int ldw, int ldc) {
#if HAS_ASYNC_LDS
  __shared__ __bf16 sA[2][BM * LSTR];
  __shared__ __bf16 sB[2][BN * LSTR];
#else
  __shared__ __bf16 sA[1][BM * LSTR];
  __shared__ __bf16 sB[1][BN * LSTR];
#endif

  const int tid  = threadIdx.x;
  const int bn   = blockIdx.x, bm = blockIdx.y;
  const int wid  = tid >> 5, lane = tid & 31;
  const int wrow = wid >> 1, wcol = wid & 1;     // waves: 4 (M) x 2 (N)
  const int lr   = lane & 15, lh = lane >> 4;

  facc_t acc[2][2];
#pragma unroll
  for (int i = 0; i < 2; ++i)
#pragma unroll
    for (int j = 0; j < 2; ++j) acc[i][j] = (facc_t){0, 0, 0, 0, 0, 0, 0, 0};

  const int nk = Kd / BK;

#if HAS_ASYNC_LDS
  // double-buffered: DMA tile k+1 while WMMA consumes tile k
  stage_tile_async(A, W, M, lda, ldw, bm, bn, 0, tid, sA[0], sB[0]);
  for (int kb = 0; kb < nk; ++kb) {
    wait_async0();        // tile kb resident in LDS
    __syncthreads();      // all waves past reads of tile kb-1
    if (kb + 1 < nk)
      stage_tile_async(A, W, M, lda, ldw, bm, bn, (kb + 1) * BK, tid,
                       sA[(kb + 1) & 1], sB[(kb + 1) & 1]);
    const __bf16* cA = sA[kb & 1];
    const __bf16* cB = sB[kb & 1];
#else
  for (int kb = 0; kb < nk; ++kb) {
    const int k0 = kb * BK;
    __syncthreads();
    // stage A tile 128x32 (two 16B chunks per thread), zero-fill past M
#pragma unroll
    for (int i = 0; i < 2; ++i) {
      int c = tid + i * 256;
      int row = c >> 2, cc = (c & 3) * 8;
      int grow = bm * BM + row;
      u32x4 v = (u32x4){0, 0, 0, 0};
      if (grow < M) {
        const __bf16* p = A + (size_t)grow * lda + k0 + cc;
        v = *(const u32x4*)p;
        if (k0 + BK < Kd) __builtin_prefetch(p + BK, 0, 0);
      }
      *(u32x4*)&sA[0][row * LSTR + cc] = v;
    }
    // stage W tile 64x32 (one 16B chunk per thread)
    {
      int row = tid >> 2, cc = (tid & 3) * 8;
      int grow = bn * BN + row;
      const __bf16* p = W + (size_t)grow * ldw + k0 + cc;
      *(u32x4*)&sB[0][row * LSTR + cc] = *(const u32x4*)p;
    }
    __syncthreads();
    const __bf16* cA = sA[0];
    const __bf16* cB = sB[0];
#endif

    FragU af[2], bf[2];
#pragma unroll
    for (int mt = 0; mt < 2; ++mt) {
      int base = (wrow * 32 + mt * 16 + lr) * LSTR + lh * 8;
      af[mt].u[0] = *(const u32x4*)&cA[base];
      af[mt].u[1] = *(const u32x4*)&cA[base + 16];
    }
#pragma unroll
    for (int nt = 0; nt < 2; ++nt) {
      int base = (wcol * 32 + nt * 16 + lr) * LSTR + lh * 8;
      bf[nt].u[0] = *(const u32x4*)&cB[base];
      bf[nt].u[1] = *(const u32x4*)&cB[base + 16];
    }
#pragma unroll
    for (int mt = 0; mt < 2; ++mt)
#pragma unroll
      for (int nt = 0; nt < 2; ++nt)
        acc[mt][nt] = __builtin_amdgcn_wmma_f32_16x16x32_bf16(
            false, af[mt].f, false, bf[nt].f, (short)0, acc[mt][nt], false, false);
  }

  // epilogue
#pragma unroll
  for (int mt = 0; mt < 2; ++mt)
#pragma unroll
    for (int nt = 0; nt < 2; ++nt) {
      const int col = bn * BN + wcol * 32 + nt * 16 + lr;
      const float bv = bias ? bias[col] : 0.f;
#pragma unroll
      for (int r = 0; r < 8; ++r) {
        const int row = bm * BM + wrow * 32 + mt * 16 + lh * 8 + r;
        if (row < M) {
          float v = acc[mt][nt][r] + bv;
          size_t idx = (size_t)row * ldc + col;
          if (MODE == 0) Cf[idx] = v;
          else if (MODE == 1) Cf[idx] = Cf[idx] + v;
          else if (MODE == 2) Cb[idx] = (__bf16)v;
          else { // exact GELU
            v = 0.5f * v * (1.f + erff(v * 0.70710678118654752f));
            Cb[idx] = (__bf16)v;
          }
        }
      }
    }
}

// ---------------- flash attention (per b,h, 128-query chunk) ----------------
__global__ __launch_bounds__(256) void attn_kernel(const __bf16* __restrict__ qkv,
                                                   const int* __restrict__ tokens,
                                                   __bf16* __restrict__ o) {
  __shared__ __bf16 sVt[64 * LSTR];        // V^T tile: [d=64][key=32] padded
  __shared__ __bf16 sP[8][16 * LSTR];      // per-wave P tile: [q=16][key=32]
  __shared__ unsigned char smask[DL];

  const int tid = threadIdx.x, wid = tid >> 5, lane = tid & 31;
  const int lr = lane & 15, lh = lane >> 4;
  const int blk = blockIdx.x;
  const int qc = blk & 3, hh = (blk >> 2) & 3, b = blk >> 4;

  const size_t rowbase = (size_t)b * DL;
  const __bf16* Q  = qkv + rowbase * 768 + hh * DHD;
  const __bf16* Kp = Q + DD;
  const __bf16* Vp = Q + 2 * DD;

  for (int l = tid; l < DL; l += 256) smask[l] = (tokens[b * DL + l] != 0);
  __syncthreads();

  const int qbase = qc * 128 + wid * 16;

  // Q fragments for d = 0..31 and 32..63, kept in registers
  FragU qa[2];
#pragma unroll
  for (int s = 0; s < 2; ++s) {
    const __bf16* p = Q + (size_t)(qbase + lr) * 768 + s * 32 + lh * 8;
    qa[s].u[0] = *(const u32x4*)p;
    qa[s].u[1] = *(const u32x4*)(p + 16);
  }

  facc_t oacc[4];
#pragma unroll
  for (int nt = 0; nt < 4; ++nt) oacc[nt] = (facc_t){0, 0, 0, 0, 0, 0, 0, 0};
  float mrow[8], srow[8];
#pragma unroll
  for (int r = 0; r < 8; ++r) { mrow[r] = -1e30f; srow[r] = 0.f; }

  const float scl = 0.125f;   // 1/sqrt(64)

  for (int kt = 0; kt < DL / 32; ++kt) {
    const int kb = kt * 32;
    __syncthreads();                 // protect sVt from previous iteration readers
    // cooperative V stage with transpose: V[32 keys][64 d] -> sVt[d][key]
    {
      int kk = tid >> 3, dc = (tid & 7) * 8;
      const __bf16* p = Vp + (size_t)(kb + kk) * 768 + dc;
      u32x4 v = *(const u32x4*)p;
      __bf16 tmp[8];
      *(u32x4*)tmp = v;
#pragma unroll
      for (int j = 0; j < 8; ++j) sVt[(dc + j) * LSTR + kk] = tmp[j];
    }
    // scores: two 16-key subtiles  S = Q K^T
    facc_t sc[2];
#pragma unroll
    for (int st = 0; st < 2; ++st) {
      sc[st] = (facc_t){0, 0, 0, 0, 0, 0, 0, 0};
#pragma unroll
      for (int s = 0; s < 2; ++s) {
        FragU kf;
        const __bf16* p = Kp + (size_t)(kb + st * 16 + lr) * 768 + s * 32 + lh * 8;
        kf.u[0] = *(const u32x4*)p;
        kf.u[1] = *(const u32x4*)(p + 16);
        sc[st] = __builtin_amdgcn_wmma_f32_16x16x32_bf16(
            false, qa[s].f, false, kf.f, (short)0, sc[st], false, false);
      }
    }
    // key-padding mask + scale (lane's column is kb + st*16 + lr)
#pragma unroll
    for (int st = 0; st < 2; ++st) {
      const bool mk = smask[kb + st * 16 + lr] != 0;
#pragma unroll
      for (int r = 0; r < 8; ++r) {
        float v = sc[st][r] * scl;
        sc[st][r] = mk ? v : -1e9f;
      }
    }
    // online softmax per row (rows distributed across 16-lane groups)
#pragma unroll
    for (int r = 0; r < 8; ++r) {
      float tm = fmaxf(sc[0][r], sc[1][r]);
#pragma unroll
      for (int m = 8; m >= 1; m >>= 1) tm = fmaxf(tm, __shfl_xor(tm, m, 32));
      const float mn = fmaxf(mrow[r], tm);
      const float rescale = __expf(mrow[r] - mn);
      const float p0 = __expf(sc[0][r] - mn);
      const float p1 = __expf(sc[1][r] - mn);
      sc[0][r] = p0; sc[1][r] = p1;
      float ts = p0 + p1;
#pragma unroll
      for (int m = 8; m >= 1; m >>= 1) ts += __shfl_xor(ts, m, 32);
      srow[r] = srow[r] * rescale + ts;
      mrow[r] = mn;
#pragma unroll
      for (int nt = 0; nt < 4; ++nt) oacc[nt][r] *= rescale;
    }
    // write P (C-layout -> LDS) so it can be regathered as an A fragment
    {
      __bf16* pw = &sP[wid][0];
#pragma unroll
      for (int st = 0; st < 2; ++st)
#pragma unroll
        for (int r = 0; r < 8; ++r)
          pw[(lh * 8 + r) * LSTR + st * 16 + lr] = (__bf16)sc[st][r];
    }
    __syncthreads();                 // sVt + sP visible
    // O += P @ V  (A = P[16,32], B = Vt rows = V columns)
    FragU pa;
    {
      int base = lr * LSTR + lh * 8;
      pa.u[0] = *(const u32x4*)&sP[wid][base];
      pa.u[1] = *(const u32x4*)&sP[wid][base + 16];
    }
#pragma unroll
    for (int nt = 0; nt < 4; ++nt) {
      FragU vf;
      int base = (nt * 16 + lr) * LSTR + lh * 8;
      vf.u[0] = *(const u32x4*)&sVt[base];
      vf.u[1] = *(const u32x4*)&sVt[base + 16];
      oacc[nt] = __builtin_amdgcn_wmma_f32_16x16x32_bf16(
          false, pa.f, false, vf.f, (short)0, oacc[nt], false, false);
    }
  }

  // normalize and write o (bf16) at [b, q, hh*64 + d]
#pragma unroll
  for (int nt = 0; nt < 4; ++nt) {
    const int d = nt * 16 + lr;
#pragma unroll
    for (int r = 0; r < 8; ++r) {
      const int q = qbase + lh * 8 + r;
      o[(rowbase + q) * DD + hh * DHD + d] = (__bf16)(oacc[nt][r] / srow[r]);
    }
  }
}

// ---------------- attention pooling over the sequence ----------------
__global__ __launch_bounds__(256) void pool_kernel(const float* __restrict__ x,
                                                   const int* __restrict__ tokens,
                                                   float* __restrict__ z,
                                                   __bf16* __restrict__ zb) {
  __shared__ float sc[DL];
  __shared__ float q[DD];
  __shared__ float red[8];
  const int b = blockIdx.x, t = threadIdx.x;

  // sequence length -> last real token index
  float cf = 0.f;
  for (int l = t; l < DL; l += 256) cf += (tokens[b * DL + l] != 0) ? 1.f : 0.f;
  cf = blockReduceSum256(cf, red);
  const int last = max((int)(cf + 0.5f) - 1, 0);

  q[t] = x[((size_t)b * DL + last) * DD + t];
  __syncthreads();

  for (int l = t; l < DL; l += 256) {
    const float* row = x + ((size_t)b * DL + l) * DD;
    float s = 0.f;
    for (int d = 0; d < DD; ++d) s += q[d] * row[d];
    sc[l] = (tokens[b * DL + l] != 0) ? s : -60000.0f;
  }
  __syncthreads();

  float pm = -3.4e38f;
  for (int l = t; l < DL; l += 256) pm = fmaxf(pm, sc[l]);
  const float bm = blockReduceMax256(pm, red);
  float ps = 0.f;
  for (int l = t; l < DL; l += 256) {
    float e = __expf(sc[l] - bm);
    sc[l] = e;
    ps += e;
  }
  const float bs = blockReduceSum256(ps, red);
  const float inv = 1.f / bs;
  __syncthreads();

  float acc = 0.f;
  for (int l = 0; l < DL; ++l) acc += sc[l] * x[((size_t)b * DL + l) * DD + t];
  acc *= inv;
  z[b * DD + t] = acc;
  zb[b * DD + t] = (__bf16)acc;
}

// ---------------- GRU cell memory update ----------------
__global__ __launch_bounds__(256) void gru_kernel(const float* __restrict__ z,
                                                  float* __restrict__ m,
                                                  const float* __restrict__ Wih,
                                                  const float* __restrict__ Whh,
                                                  const float* __restrict__ bih,
                                                  const float* __restrict__ bhh) {
  __shared__ float gi[3 * DD], gh[3 * DD], zs[DD], ms[DD];
  const int b = blockIdx.x, t = threadIdx.x;
  zs[t] = z[b * DD + t];
  ms[t] = m[b * DD + t];
  __syncthreads();
#pragma unroll
  for (int j = 0; j < 3; ++j) {
    const int c = t + j * DD;
    const float* wi = Wih + (size_t)c * DD;
    const float* wh = Whh + (size_t)c * DD;
    float a1 = bih[c], a2 = bhh[c];
    for (int d = 0; d < DD; ++d) { a1 += zs[d] * wi[d]; a2 += ms[d] * wh[d]; }
    gi[c] = a1; gh[c] = a2;
  }
  __syncthreads();
  const float r = 1.f / (1.f + __expf(-(gi[t] + gh[t])));
  const float u = 1.f / (1.f + __expf(-(gi[t + DD] + gh[t + DD])));
  const float n = tanhf(gi[t + 2 * DD] + r * gh[t + 2 * DD]);
  m[b * DD + t] = (1.f - u) * n + u * ms[t];
}

// ---------------- host-side orchestration ----------------
extern "C" void kernel_launch(void* const* d_in, const int* in_sizes, int n_in,
                              void* d_out, int out_size, void* d_ws, size_t ws_size,
                              hipStream_t stream) {
  (void)in_sizes; (void)n_in; (void)out_size; (void)ws_size;

  const int*   tokens  = (const int*)  d_in[0];
  // d_in[1] (bool mask) intentionally unused: mask == (tokens != 0) by construction
  const float* tok_emb = (const float*)d_in[2];
  const float* pos_emb = (const float*)d_in[3];
  const float* eg   = (const float*)d_in[4];
  const float* ebb  = (const float*)d_in[5];
  const float* ln1g = (const float*)d_in[6];
  const float* ln1b = (const float*)d_in[7];
  const float* Wqkv = (const float*)d_in[8];
  const float* bqkv = (const float*)d_in[9];
  const float* Wo   = (const float*)d_in[10];
  const float* bo   = (const float*)d_in[11];
  const float* ln2g = (const float*)d_in[12];
  const float* ln2b = (const float*)d_in[13];
  const float* W1   = (const float*)d_in[14];
  const float* b1   = (const float*)d_in[15];
  const float* W2   = (const float*)d_in[16];
  const float* b2   = (const float*)d_in[17];
  const float* gWih = (const float*)d_in[18];
  const float* gWhh = (const float*)d_in[19];
  const float* gbih = (const float*)d_in[20];
  const float* gbhh = (const float*)d_in[21];
  const float* clsW = (const float*)d_in[22];
  const float* clsb = (const float*)d_in[23];

  char* ws = (char*)d_ws;
  size_t off = 0;
  auto take = [&](size_t bytes) -> char* {
    char* p = ws + off;
    off = (off + bytes + 255) & ~(size_t)255;
    return p;
  };
  float*  x    = (float*) take((size_t)DM * DD * 4);
  __bf16* hbuf = (__bf16*)take((size_t)DM * DD * 2);
  __bf16* qkvB = (__bf16*)take((size_t)DM * 3 * DD * 2);
  __bf16* obuf = (__bf16*)take((size_t)DM * DD * 2);
  __bf16* ffB  = (__bf16*)take((size_t)DM * DFF * 2);
  float*  z    = (float*) take((size_t)DB * DD * 4);
  __bf16* zb   = (__bf16*)take((size_t)DB * DD * 2);
  float*  mmem = (float*) take((size_t)DB * DD * 4);
  __bf16* wqkvB = (__bf16*)take((size_t)DK * 3 * DD * DD * 2);
  __bf16* woB   = (__bf16*)take((size_t)DK * DD * DD * 2);
  __bf16* w1B   = (__bf16*)take((size_t)DK * DFF * DD * 2);
  __bf16* w2B   = (__bf16*)take((size_t)DK * DD * DFF * 2);
  __bf16* clsB  = (__bf16*)take((size_t)DV * DD * 2);

  auto cvt = [&](const float* in, __bf16* out, long long n) {
    int grid = (int)((n + 255) / 256);
    if (grid > 4096) grid = 4096;
    cvt_bf16_kernel<<<grid, 256, 0, stream>>>(in, out, n);
  };
  cvt(Wqkv, wqkvB, (long long)DK * 3 * DD * DD);
  cvt(Wo,   woB,   (long long)DK * DD * DD);
  cvt(W1,   w1B,   (long long)DK * DFF * DD);
  cvt(W2,   w2B,   (long long)DK * DD * DFF);
  cvt(clsW, clsB,  (long long)DV * DD);

  (void)hipMemsetAsync(mmem, 0, (size_t)DB * DD * 4, stream);

  embed_ln_kernel<<<DM, 256, 0, stream>>>(tokens, tok_emb, pos_emb, eg, ebb, x);

  const dim3 gQKV(3 * DD / BN, DM / BM);
  const dim3 gD  (DD / BN,     DM / BM);
  const dim3 gFF (DFF / BN,    DM / BM);
  const dim3 gCLS(DV / BN, 1);

  for (int e = 0; e < DK; ++e) {
    // h = LN1(x)
    ln_bf16_kernel<<<DM, 256, 0, stream>>>(x, ln1g + e * DD, ln1b + e * DD, hbuf);
    // qkv = h @ Wqkv^T + bqkv   (bf16 out)
    gemm_wmma_kernel<2><<<gQKV, 256, 0, stream>>>(
        hbuf, wqkvB + (size_t)e * 3 * DD * DD, bqkv + e * 3 * DD,
        nullptr, qkvB, DM, 3 * DD, DD, DD, DD, 3 * DD);
    // attention
    attn_kernel<<<DB * DH * 4, 256, 0, stream>>>(qkvB, tokens, obuf);
    // x += o @ Wo^T + bo
    gemm_wmma_kernel<1><<<gD, 256, 0, stream>>>(
        obuf, woB + (size_t)e * DD * DD, bo + e * DD,
        x, nullptr, DM, DD, DD, DD, DD, DD);
    // h = LN2(x)
    ln_bf16_kernel<<<DM, 256, 0, stream>>>(x, ln2g + e * DD, ln2b + e * DD, hbuf);
    // ff = gelu(h @ W1^T + b1)  (bf16 out)
    gemm_wmma_kernel<3><<<gFF, 256, 0, stream>>>(
        hbuf, w1B + (size_t)e * DFF * DD, b1 + e * DFF,
        nullptr, ffB, DM, DFF, DD, DD, DD, DFF);
    // x += ff @ W2^T + b2
    gemm_wmma_kernel<1><<<gD, 256, 0, stream>>>(
        ffB, w2B + (size_t)e * DD * DFF, b2 + e * DD,
        x, nullptr, DM, DD, DFF, DFF, DFF, DD);
    // z = pool(x), m = GRU(z, m)
    pool_kernel<<<DB, 256, 0, stream>>>(x, tokens, z, zb);
    gru_kernel<<<DB, 256, 0, stream>>>(z, mmem, gWih, gWhh, gbih, gbhh);
  }

  // logits = pool(x) @ cls_W^T + cls_b  (zb from the last pool == final pool)
  gemm_wmma_kernel<0><<<gCLS, 256, 0, stream>>>(
      zb, clsB, clsb, (float*)d_out, nullptr, DB, DV, DD, DD, DD, DV);
}
